// CharRNN_36558761624300
// MI455X (gfx1250) — matrix operand order repaired and checked by reading, not compiled, not run on hardware
//
#include <hip/hip_runtime.h>
#include <hip/hip_bf16.h>

// CharRNN fused forward for MI455X (gfx1250, wave32, WMMA).
// Shapes: VOCAB=96, E=32, H=128, B=512, T=512.
// One workgroup = 16 batch rows for all T steps; 8 waves of 32.
// Wave w computes h columns [16w,16w+16); waves 0..5 also compute the
// 96 logit columns. Weights live in registers as bf16-pair B-layout tiles,
// h_t lives in double-buffered LDS (1 barrier/step).
// Latency-bound on the serial T chain -> branchless hw tanh + split WMMA
// accumulator chains to shorten the per-step dependency depth.

#define VOCAB 96
#define EDIM  32
#define HDIM  128
#define BSZ   512
#define TSZ   512

typedef __attribute__((ext_vector_type(16))) __bf16 v16bf;
typedef __attribute__((ext_vector_type(8)))  float  v8f;
typedef __attribute__((ext_vector_type(2)))  __bf16 v2bf;

struct R8 { unsigned u[8]; };

__device__ __forceinline__ unsigned pk(float a, float b) {
    v2bf p; p[0] = (__bf16)a; p[1] = (__bf16)b;   // K even -> [15:0], K+1 -> [31:16]
    return __builtin_bit_cast(unsigned, p);
}
__device__ __forceinline__ v16bf as16(const R8& r) {
    return __builtin_bit_cast(v16bf, r);
}
__device__ __forceinline__ v8f wmma_bf16(const R8& a, const R8& b, v8f c) {
    return __builtin_amdgcn_wmma_f32_16x16x32_bf16(
        /*neg_a=*/false, as16(a), /*neg_b=*/false, as16(b),
        /*c_mod=*/(short)0, c, /*reuse_a=*/false, /*reuse_b=*/false);
}

// Branchless tanh: prefer the CDNA5 V_TANH_F32 transcendental; fallback uses
// exp2/rcp TRANS ops only (no EXEC-divergent branches on the serial path).
__device__ __forceinline__ float fast_tanh(float x) {
#if __has_builtin(__builtin_amdgcn_tanhf)
    return __builtin_amdgcn_tanhf(x);
#elif __has_builtin(__builtin_amdgcn_tanh_f32)
    return __builtin_amdgcn_tanh_f32(x);
#else
    const float a = __builtin_fabsf(x);
    const float e = __builtin_amdgcn_exp2f(-2.8853900817779268f * a); // e^(2|x|)^-1
    const float r = __builtin_amdgcn_rcpf(1.0f + e);
    const float t = (1.0f - e) * r;                 // tanh(|x|), e in (0,1]
    return __builtin_copysignf(t, x);
#endif
}

__global__ __launch_bounds__(256)
void rnn_fused(const int*   __restrict__ x,
               const float* __restrict__ emb,
               const float* __restrict__ Wxh,
               const float* __restrict__ Whh,
               const float* __restrict__ bh,
               const float* __restrict__ fcW,
               const float* __restrict__ fcb,
               float*       __restrict__ out)
{
    __shared__ unsigned emb_lds[VOCAB * EDIM / 2];          // bf16 pairs, 6 KB
    __shared__ int      xs[16 * TSZ];                       // 32 KB token ids
    __shared__ __align__(16) __bf16 hbuf[2][16][HDIM];      // 8 KB, double-buffered h

    const int tid   = threadIdx.x;
    const int lane  = tid & 31;
    const int wv    = tid >> 5;        // wave 0..7
    const int lm    = lane & 15;       // row (A) / col (B,C,D) within tile
    const int hi    = lane >> 4;       // half-wave select
    const int bbase = blockIdx.x * 16; // batch-row base
    const int ncol  = wv * 16 + lm;    // this wave's H column (also logits col for wv<6)

    // ---- stage embedding table (f32 -> packed bf16) and token ids into LDS ----
    for (int i = tid; i < VOCAB * EDIM / 2; i += 256)
        emb_lds[i] = pk(emb[2 * i], emb[2 * i + 1]);
    for (int i = tid; i < 16 * TSZ; i += 256)
        xs[i] = x[(bbase + (i >> 9)) * TSZ + (i & (TSZ - 1))];

    // Per-lane K offsets for A / B VGPR layouts (CDNA5 ISA §7.12.2).
    int kA[8], kB[8];
#pragma unroll
    for (int v = 0; v < 8; ++v) {
        kA[v] = ((v & 3) << 1) + (hi << 3) + ((v >> 2) << 4); // A 16x32 bf16
        kB[v] = ((v & 3) << 1) + (hi << 3) + ((v >> 2) << 4); // B 32x16 bf16
    }

    // ---- preload all B-layout weight tiles into registers (bf16 pairs) ----
    R8 Bxh;                       // Wxh slice: K=0..31, N=ncol
#pragma unroll
    for (int v = 0; v < 8; ++v) {
        const int k0 = kB[v];
        Bxh.u[v] = pk(Wxh[k0 * HDIM + ncol], Wxh[(k0 + 1) * HDIM + ncol]);
    }
    R8 Bhh[4];                    // Whh slices: 4 K-tiles of 32
#pragma unroll
    for (int kt = 0; kt < 4; ++kt)
#pragma unroll
        for (int v = 0; v < 8; ++v) {
            const int k0 = kt * 32 + kB[v];
            Bhh[kt].u[v] = pk(Whh[k0 * HDIM + ncol], Whh[(k0 + 1) * HDIM + ncol]);
        }
    R8 Bfc[4];                    // fc_W slices (waves 0..5 only)
    float fcbv = 0.0f;
    if (wv < 6) {
#pragma unroll
        for (int kt = 0; kt < 4; ++kt)
#pragma unroll
            for (int v = 0; v < 8; ++v) {
                const int k0 = kt * 32 + kB[v];
                Bfc[kt].u[v] = pk(fcW[k0 * VOCAB + ncol], fcW[(k0 + 1) * VOCAB + ncol]);
            }
        fcbv = fcb[ncol];
    } else {
#pragma unroll
        for (int kt = 0; kt < 4; ++kt)
#pragma unroll
            for (int v = 0; v < 8; ++v) Bfc[kt].u[v] = 0u;
    }
    const float bhv = bh[ncol];

    // Per-lane running output pointer: row offsets r*T*V fold into the
    // signed 24-bit instruction offset of each global_store_b32.
    float* orow = out + (size_t)(bbase + 8 * hi) * TSZ * VOCAB + ncol;

    __syncthreads();

    // h_{-1} = 0 in A-layout registers
    R8 Ah[4];
#pragma unroll
    for (int kt = 0; kt < 4; ++kt)
#pragma unroll
        for (int v = 0; v < 8; ++v) Ah[kt].u[v] = 0u;

    // prefetch A_x for t = 0 (embedding rows of this tile's 16 tokens)
    R8 Ax;
    {
        const int tok = xs[lm * TSZ + 0];
#pragma unroll
        for (int v = 0; v < 8; ++v)
            Ax.u[v] = emb_lds[tok * (EDIM / 2) + (kA[v] >> 1)];
    }

    for (int t = 0; t < TSZ; ++t) {
        // ---- h_t = tanh(bh + x_emb@Wxh + h_{t-1}@Whh) ----
        // Two accumulator chains (depth 3 + depth 2) merged with VALU adds
        // to shorten the serial WMMA dependency per step.
        v8f acc0, acc1;
#pragma unroll
        for (int r = 0; r < 8; ++r) { acc0[r] = bhv; acc1[r] = 0.0f; }
        acc0 = wmma_bf16(Ax, Bxh, acc0);
        acc0 = wmma_bf16(Ah[0], Bhh[0], acc0);
        acc1 = wmma_bf16(Ah[2], Bhh[2], acc1);
        acc0 = wmma_bf16(Ah[1], Bhh[1], acc0);
        acc1 = wmma_bf16(Ah[3], Bhh[3], acc1);

        const int buf = t & 1;
#pragma unroll
        for (int r = 0; r < 8; ++r) {
            const float hv = fast_tanh(acc0[r] + acc1[r]);
            hbuf[buf][r + 8 * hi][ncol] = (__bf16)hv;   // C layout -> row-major LDS
        }

        // pipeline: next step's embedding gather is independent of h_t
        if (t + 1 < TSZ) {
            const int tok = xs[lm * TSZ + t + 1];
#pragma unroll
            for (int v = 0; v < 8; ++v)
                Ax.u[v] = emb_lds[tok * (EDIM / 2) + (kA[v] >> 1)];
        }

        __syncthreads();   // one barrier per step (double-buffered hbuf)

        // reload full h_t in A-layout (used for logits now and recurrence next step)
#pragma unroll
        for (int kt = 0; kt < 4; ++kt)
#pragma unroll
            for (int v = 0; v < 8; ++v) {
                const int k0 = kt * 32 + kA[v];
                Ah[kt].u[v] = *(const unsigned*)&hbuf[buf][lm][k0];
            }

        // ---- logits tile: h_t @ fc_W + fc_b, overlapped with next-step latency ----
        if (wv < 6) {
            v8f o0, o1;
#pragma unroll
            for (int r = 0; r < 8; ++r) { o0[r] = fcbv; o1[r] = 0.0f; }
            o0 = wmma_bf16(Ah[0], Bfc[0], o0);
            o1 = wmma_bf16(Ah[2], Bfc[2], o1);
            o0 = wmma_bf16(Ah[1], Bfc[1], o0);
            o1 = wmma_bf16(Ah[3], Bfc[3], o1);
#pragma unroll
            for (int r = 0; r < 8; ++r)
                orow[(size_t)r * TSZ * VOCAB] = o0[r] + o1[r];
        }
        orow += VOCAB;
    }
}

extern "C" void kernel_launch(void* const* d_in, const int* in_sizes, int n_in,
                              void* d_out, int out_size, void* d_ws, size_t ws_size,
                              hipStream_t stream) {
    (void)in_sizes; (void)n_in; (void)out_size; (void)d_ws; (void)ws_size;
    const int*   x    = (const int*)  d_in[0];  // token ids (int32 per harness convention)
    const float* emb  = (const float*)d_in[1];
    const float* Wxh  = (const float*)d_in[2];
    const float* Whh  = (const float*)d_in[3];
    const float* bh   = (const float*)d_in[4];
    const float* fcW  = (const float*)d_in[5];
    const float* fcb  = (const float*)d_in[6];
    float* out = (float*)d_out;

    rnn_fused<<<BSZ / 16, 256, 0, stream>>>(x, emb, Wxh, Whh, bh, fcW, fcb, out);
}
